// Set2SetEncoder_29420525978327
// MI455X (gfx1250) — compile-verified, matrix operand order checked by reading, not run on hardware
//
#include <hip/hip_runtime.h>

// ---------------------------------------------------------------------------
// Set2Set encoder for MI455X (gfx1250, wave32, WMMA).
//   B=256, N=1024, D=128, HID=256.
// Strategy:
//   * One-time prep: x -> f16 (64MB, L2-resident), weights -> f16 packed in
//     WMMA B-fragment lane order, bias = b_ih + b_hh.
//   * Main persistent kernel: 64 workgroups x 512 threads (16 waves).
//     Workgroup (tile,gidx): tile = batch tile of 16 rows, gidx = quarter of
//     the set dimension (256 of 1024 set elements).
//     Gates GEMM (M=16 batch) via v_wmma_f32_16x16x32_f16 (K-slab order is
//     rotated by step so the L2-resident weight loads are NOT hoisted out of
//     the 1024-step loop -> no register spills), LSTM pointwise with c-state
//     in VGPRs, attention pass 1 on VALU from L2-resident f16 x with
//     global_prefetch, attention pass 2 double-buffered through LDS with
//     GLOBAL_LOAD_ASYNC_TO_LDS_B128 + s_wait_asynccnt (fallback: direct
//     global loads), cross-workgroup softmax/r combine via global scratch +
//     atomic-counter group barrier (monotonic count, s_sleep spin).
//   * Final linear: WMMA (16x256)@(256x256 HID) per tile, gidx==0 stores.
// ---------------------------------------------------------------------------

typedef __attribute__((ext_vector_type(16))) _Float16 v16h;
typedef __attribute__((ext_vector_type(8)))  _Float16 v8h;
typedef __attribute__((ext_vector_type(4)))  _Float16 v4h;
typedef __attribute__((ext_vector_type(8)))  float    v8f;
typedef __attribute__((ext_vector_type(4)))  float    v4f;
typedef int v4i __attribute__((vector_size(4 * sizeof(int))));

#define B_    256
#define N_    1024
#define D_    128
#define HID_  256
#define G4    4            // workgroups per batch tile (split of set dim)
#define NLOC  (N_ / G4)    // 256 set elements per workgroup
#define TPB   512          // 16 waves of 32

#if defined(__HIP_DEVICE_COMPILE__) && defined(__has_builtin)
#if __has_builtin(__builtin_amdgcn_global_load_async_to_lds_b128) && \
    __has_builtin(__builtin_amdgcn_s_wait_asynccnt)
#define HAS_ASYNC_LDS 1
#endif
#endif

#ifdef HAS_ASYNC_LDS
#define AS1 __attribute__((address_space(1)))
#define AS3 __attribute__((address_space(3)))
#endif

// ---- workspace layout (bytes, all regions 256B aligned) -------------------
constexpr size_t X16_OFF  = 0;                                   // 256*1024*128 f16
constexpr size_t WIH_OFF  = (size_t)B_ * N_ * D_ * 2;            // 67108864
constexpr size_t WHH_OFF  = WIH_OFF + (size_t)512 * 256 * 2;     // +262144
constexpr size_t WLIN_OFF = WHH_OFF + (size_t)512 * 128 * 2;     // +131072
constexpr size_t BIAS_OFF = WLIN_OFF + (size_t)256 * 256 * 2;    // +131072
constexpr size_t BLIN_OFF = BIAS_OFF + 512 * 4;
constexpr size_t MS_OFF   = BLIN_OFF + 256 * 4;                  // 16*4*16*2 f32
constexpr size_t RP_OFF   = MS_OFF + 16 * 4 * 16 * 2 * 4;        // 16*4*16*128 f32
constexpr size_t CNT_OFF  = RP_OFF + (size_t)16 * 4 * 16 * 128 * 4;

__device__ __forceinline__ float sigf(float x) { return 1.f / (1.f + __expf(-x)); }
__device__ __forceinline__ float tanhf_(float x) {
    float t = __expf(-2.f * x);
    return (1.f - t) / (1.f + t);
}

__device__ __forceinline__ v8f wmma16(v16h a, v16h b, v8f c) {
    // (neg_a, A, neg_b, B, c_mod, C, reuse_a, reuse_b)
    return __builtin_amdgcn_wmma_f32_16x16x32_f16(false, a, false, b, (short)0, c,
                                                  false, false);
}

// group barrier among the G4 workgroups of one batch tile: monotonic counter
__device__ __forceinline__ void group_barrier(unsigned* cnt, unsigned expected) {
    __threadfence();
    __syncthreads();
    if (threadIdx.x == 0) {
        atomicAdd(cnt, 1u);
        while (__hip_atomic_load(cnt, __ATOMIC_RELAXED, __HIP_MEMORY_SCOPE_AGENT) <
               expected) {
            __builtin_amdgcn_s_sleep(1);
        }
    }
    __syncthreads();
    __threadfence();   // executed by all threads -> every CU's L0 invalidated
}

// ---------------------------------------------------------------------------
// Prep: x fp32 -> f16
__global__ void k_prep_x(const float* __restrict__ x, _Float16* __restrict__ x16,
                         int n) {
    for (int i = blockIdx.x * blockDim.x + threadIdx.x; i < n;
         i += gridDim.x * blockDim.x)
        x16[i] = (_Float16)x[i];
}

// Prep: pack weights into WMMA B-fragment lane order, biases, barrier counters.
// Fragment map (16-bit, 32x16 K x N tile): lane = n + 16*((k>>3)&1),
// pos(0..15) = (k&7) + 8*(k>>4). Inverse used here.
__global__ void k_prep_w(const float* __restrict__ Wih, const float* __restrict__ Whh,
                         const float* __restrict__ Wlin, const float* __restrict__ bih,
                         const float* __restrict__ bhh, const float* __restrict__ blin_in,
                         _Float16* __restrict__ wih, _Float16* __restrict__ whh,
                         _Float16* __restrict__ wlin, float* __restrict__ bias,
                         float* __restrict__ blin, unsigned* __restrict__ cnt) {
    const int total = 131072 + 65536 + 65536 + 512 + 256 + 16;
    for (int s = blockIdx.x * blockDim.x + threadIdx.x; s < total;
         s += gridDim.x * blockDim.x) {
        if (s < 131072) {                       // W_ih^T : K=256, N=512 -> [nt32][ks8][lane][pos]
            int pos = s & 15, lane = (s >> 4) & 31, kn = s >> 9;
            int ks = kn & 7, nt = kn >> 3;
            int n = nt * 16 + (lane & 15);
            int k = ks * 32 + ((pos & 7) | (((lane >> 4) & 1) << 3) | ((pos >> 3) << 4));
            wih[s] = (_Float16)Wih[n * 256 + k];
        } else if (s < 196608) {                // W_hh^T : K=128, N=512 -> [nt32][ks4][lane][pos]
            int t = s - 131072;
            int pos = t & 15, lane = (t >> 4) & 31, kn = t >> 9;
            int ks = kn & 3, nt = kn >> 2;
            int n = nt * 16 + (lane & 15);
            int k = ks * 32 + ((pos & 7) | (((lane >> 4) & 1) << 3) | ((pos >> 3) << 4));
            whh[t] = (_Float16)Whh[n * 128 + k];
        } else if (s < 262144) {                // W_lin^T : K=256, N=256 -> [nt16][ks8][lane][pos]
            int t = s - 196608;
            int pos = t & 15, lane = (t >> 4) & 31, kn = t >> 9;
            int ks = kn & 7, nt = kn >> 3;
            int n = nt * 16 + (lane & 15);
            int k = ks * 32 + ((pos & 7) | (((lane >> 4) & 1) << 3) | ((pos >> 3) << 4));
            wlin[t] = (_Float16)Wlin[n * 256 + k];
        } else if (s < 262656) {
            int g = s - 262144;
            bias[g] = bih[g] + bhh[g];
        } else if (s < 262912) {
            int g = s - 262656;
            blin[g] = blin_in[g];
        } else {
            cnt[s - 262912] = 0u;               // group-barrier counters
        }
    }
}

// ---------------------------------------------------------------------------
__global__ __launch_bounds__(TPB, 1) void k_main(
    const _Float16* __restrict__ x16, const _Float16* __restrict__ wih,
    const _Float16* __restrict__ whh, const _Float16* __restrict__ wlin,
    const float* __restrict__ bias, const float* __restrict__ blin,
    float* __restrict__ ms,      // [16][4][16][2]  (m, s) partials
    float* __restrict__ rpart,   // [16][4][16][128] partial r
    unsigned* __restrict__ cnt,  // [16] group counters
    float* __restrict__ out)     // [256][256]
{
    // gates (16x512 f32) and e-scores (16x256 f32) have disjoint lifetimes -> overlay
    __shared__ float smem_ge[16 * 512];                 // 32 KB
    __shared__ __align__(16) float q_s[16][128];        // 8 KB  (q = h, f32)
    __shared__ __align__(32) _Float16 qsf[8 * 32 * 16]; // 8 KB  q_star A-fragments (K=0..255)
#ifdef HAS_ASYNC_LDS
    __shared__ __align__(16) _Float16 xstage[16][2][8 * 128]; // 64 KB x double-buffer
#endif

#define GATES(b, g) smem_ge[(b) * 512 + (g)]
#define ES(b, nn)   smem_ge[(b) * 256 + (nn)]

    const int tid  = threadIdx.x;
    const int wave = tid >> 5;
    const int lane = tid & 31;
    const int tile = blockIdx.x >> 2;   // batch tile (16 rows)
    const int gidx = blockIdx.x & 3;    // quarter of the set dim
    const int b    = wave;              // wave w owns batch row b=w of the tile
    const int bg   = tile * 16 + b;     // global batch row

    for (int i = tid; i < 16 * 512; i += TPB) smem_ge[i] = 0.f;
    for (int i = tid; i < 16 * 128; i += TPB) ((float*)q_s)[i] = 0.f;
    for (int i = tid; i < 8 * 32 * 16; i += TPB) qsf[i] = (_Float16)0.f;
    float creg[4] = {0.f, 0.f, 0.f, 0.f};
    __syncthreads();

    const int row = lane >> 4, col = lane & 15;
    unsigned* mycnt = cnt + tile;

    for (int step = 0; step < N_; ++step) {
        // ---------------- gates GEMM: (16 x 384K) via WMMA ----------------
        // One gate tile at a time (limits live B-fragments); K-slab order is
        // rotated by `step` so weight loads are step-dependent and cannot be
        // hoisted (and spilled) out of the 1024-step loop.
#pragma unroll 1
        for (int half = 0; half < 2; ++half) {
            const int nt = wave + 16 * half;    // this wave's gate tile (of 32)
            v8f acc = {};
#pragma unroll
            for (int ks = 0; ks < 8; ++ks) {    // q_star part, K = 0..255
                int k2 = (ks + step) & 7;
                v16h a  = *(const v16h*)&qsf[(k2 * 32 + lane) * 16];
                v16h bw = *(const v16h*)(wih + ((size_t)(nt * 8 + k2) * 32 + lane) * 16);
                acc = wmma16(a, bw, acc);
            }
#pragma unroll
            for (int ks = 0; ks < 4; ++ks) {    // h part, K = 0..127 (h == q_star[:,0:128])
                int k2 = (ks + step) & 3;
                v16h a  = *(const v16h*)&qsf[(k2 * 32 + lane) * 16];
                v16h bw = *(const v16h*)(whh + ((size_t)(nt * 4 + k2) * 32 + lane) * 16);
                acc = wmma16(a, bw, acc);
            }
            // C/D layout: lane<16 -> n=lane, m=i ; lane>=16 -> n=lane-16, m=i+8
#pragma unroll
            for (int i = 0; i < 8; ++i)
                GATES(i + 8 * row, nt * 16 + col) = acc[i];
        }
        __syncthreads();

        // ---------------- LSTM pointwise (wave b; c in VGPRs) -------------
#pragma unroll
        for (int j = 0; j < 4; ++j) {
            int d = j * 32 + lane;
            float iv = GATES(b, d)       + bias[d];
            float fv = GATES(b, 128 + d) + bias[128 + d];
            float gv = GATES(b, 256 + d) + bias[256 + d];
            float ov = GATES(b, 384 + d) + bias[384 + d];
            float cc = sigf(fv) * creg[j] + sigf(iv) * tanhf_(gv);
            creg[j]  = cc;
            float h  = sigf(ov) * tanhf_(cc);
            q_s[b][d] = h;
            // scatter into A-fragment order, K = d
            int kk = d & 31, ks = d >> 5;
            int fl = b + 16 * ((kk >> 3) & 1);
            int pos = (kk & 7) + 8 * (kk >> 4);
            qsf[(ks * 32 + fl) * 16 + pos] = (_Float16)h;
        }
        __syncthreads();   // GATES region now free -> reused as ES

        // ---------------- attention pass 1: e = x.q, local softmax --------
        float m_loc = -3.4e38f;
        float evals[8];
#pragma unroll
        for (int it = 0; it < 8; ++it) {
            int nl = it * 32 + lane;
            const _Float16* xp =
                x16 + ((size_t)bg * N_ + (size_t)gidx * NLOC + nl) * D_;
            __builtin_prefetch(xp + 32 * D_, 0, 0);   // next it's row -> global_prefetch
            float e = 0.f;
#pragma unroll
            for (int dch = 0; dch < D_; dch += 8) {
                v8h xv = *(const v8h*)(xp + dch);
                v4f q0 = *(const v4f*)&q_s[b][dch];
                v4f q1 = *(const v4f*)&q_s[b][dch + 4];
                e += (float)xv[0] * q0[0];
                e += (float)xv[1] * q0[1];
                e += (float)xv[2] * q0[2];
                e += (float)xv[3] * q0[3];
                e += (float)xv[4] * q1[0];
                e += (float)xv[5] * q1[1];
                e += (float)xv[6] * q1[2];
                e += (float)xv[7] * q1[3];
            }
            ES(b, nl) = e;
            evals[it] = e;
            m_loc = fmaxf(m_loc, e);
        }
        for (int off = 16; off; off >>= 1)
            m_loc = fmaxf(m_loc, __shfl_xor(m_loc, off, 32));
        float s_loc = 0.f;
#pragma unroll
        for (int it = 0; it < 8; ++it) s_loc += __expf(evals[it] - m_loc);
        for (int off = 16; off; off >>= 1) s_loc += __shfl_xor(s_loc, off, 32);
        if (lane == 0) {
            float2 p;
            p.x = m_loc;
            p.y = s_loc;
            *(float2*)(ms + ((size_t)(tile * 4 + gidx) * 16 + b) * 2) = p;
        }
        group_barrier(mycnt, (unsigned)(8 * step + 4));

        // ---------------- combine softmax across the 4 workgroups ---------
        float M = -3.4e38f;
#pragma unroll
        for (int g = 0; g < 4; ++g)
            M = fmaxf(M, ms[((size_t)(tile * 4 + g) * 16 + b) * 2 + 0]);
        float S = 0.f;
#pragma unroll
        for (int g = 0; g < 4; ++g) {
            float mg = ms[((size_t)(tile * 4 + g) * 16 + b) * 2 + 0];
            float sg = ms[((size_t)(tile * 4 + g) * 16 + b) * 2 + 1];
            S += sg * __expf(mg - M);
        }
        float invS = 1.f / S;
#pragma unroll
        for (int it = 0; it < 8; ++it) {
            int nl = it * 32 + lane;
            ES(b, nl) = __expf(evals[it] - M) * invS;   // a[n]
        }

        // ---------------- attention pass 2: partial r = a^T x -------------
        float r0 = 0.f, r1 = 0.f, r2 = 0.f, r3 = 0.f;
        const _Float16* xslab = x16 + ((size_t)bg * N_ + (size_t)gidx * NLOC) * D_;
#ifdef HAS_ASYNC_LDS
        // Double-buffered async global->LDS staging of 8-row (2KB) chunks of x.
        // Async loads complete in order; ASYNCcnt paces the pipeline.
        {
            _Float16* st = &xstage[wave][0][0];
            auto stage = [&](int buf, int ch) {
                const _Float16* g = xslab + (size_t)ch * 8 * D_;
                _Float16* l = st + buf * (8 * 128);
#pragma unroll
                for (int t = 0; t < 4; ++t) {
                    __builtin_amdgcn_global_load_async_to_lds_b128(
                        (AS1 v4i*)(AS1 void*)(g + t * 256 + lane * 8),
                        (AS3 v4i*)(AS3 void*)(l + t * 256 + lane * 8), 0, 0);
                }
            };
            stage(0, 0);
#pragma unroll 1
            for (int ch = 0; ch < NLOC / 8; ++ch) {
                int buf = ch & 1;
                if (ch + 1 < NLOC / 8) {
                    stage(buf ^ 1, ch + 1);
                    __builtin_amdgcn_s_wait_asynccnt(4);   // chunk ch landed
                } else {
                    __builtin_amdgcn_s_wait_asynccnt(0);   // last chunk landed
                }
                const _Float16* l = st + buf * (8 * 128);
#pragma unroll
                for (int rr = 0; rr < 8; ++rr) {
                    float a = ES(b, ch * 8 + rr);
                    v4h xv = *(const v4h*)(l + rr * 128 + lane * 4);
                    r0 += a * (float)xv[0];
                    r1 += a * (float)xv[1];
                    r2 += a * (float)xv[2];
                    r3 += a * (float)xv[3];
                }
            }
        }
#else
        {
            const _Float16* xb = xslab + lane * 4;
#pragma unroll 4
            for (int nl = 0; nl < NLOC; ++nl) {
                float a = ES(b, nl);
                v4h xv = *(const v4h*)(xb + (size_t)nl * D_);
                __builtin_prefetch(xb + (size_t)(nl + 8) * D_, 0, 0);
                r0 += a * (float)xv[0];
                r1 += a * (float)xv[1];
                r2 += a * (float)xv[2];
                r3 += a * (float)xv[3];
            }
        }
#endif
        {
            v4f rv = {r0, r1, r2, r3};
            *(v4f*)(rpart + ((size_t)(tile * 4 + gidx) * 16 + b) * 128 + lane * 4) = rv;
        }
        group_barrier(mycnt, (unsigned)(8 * step + 8));

        // ---------------- combine r, write q_star r-fragments -------------
#pragma unroll
        for (int j = 0; j < 4; ++j) {
            int d = j * 32 + lane;
            float r = 0.f;
#pragma unroll
            for (int g = 0; g < 4; ++g)
                r += rpart[((size_t)(tile * 4 + g) * 16 + b) * 128 + d];
            int k = 128 + d;
            int kk = k & 31, ks = k >> 5;
            int fl = b + 16 * ((kk >> 3) & 1);
            int pos = (kk & 7) + 8 * (kk >> 4);
            qsf[(ks * 32 + fl) * 16 + pos] = (_Float16)r;
        }
        __syncthreads();
    }

    // ---------------- final linear: out = q_star @ W_lin^T + b_lin --------
    v8f acc = {};
#pragma unroll
    for (int ks = 0; ks < 8; ++ks) {
        v16h a  = *(const v16h*)&qsf[(ks * 32 + lane) * 16];
        v16h bw = *(const v16h*)(wlin + ((size_t)(wave * 8 + ks) * 32 + lane) * 16);
        acc = wmma16(a, bw, acc);
    }
    if (gidx == 0) {
#pragma unroll
        for (int i = 0; i < 8; ++i) {
            int bb = tile * 16 + i + 8 * row;
            int hh = wave * 16 + col;
            out[(size_t)bb * HID_ + hh] = acc[i] + blin[hh];
        }
    }
#undef GATES
#undef ES
}

// ---------------------------------------------------------------------------
extern "C" void kernel_launch(void* const* d_in, const int* in_sizes, int n_in,
                              void* d_out, int out_size, void* d_ws, size_t ws_size,
                              hipStream_t stream) {
    (void)in_sizes; (void)n_in; (void)out_size; (void)ws_size;
    const float* x    = (const float*)d_in[0];
    const float* Wih  = (const float*)d_in[1];
    const float* Whh  = (const float*)d_in[2];
    const float* bih  = (const float*)d_in[3];
    const float* bhh  = (const float*)d_in[4];
    const float* Wlin = (const float*)d_in[5];
    const float* blin = (const float*)d_in[6];

    char* ws = (char*)d_ws;
    _Float16* x16   = (_Float16*)(ws + X16_OFF);
    _Float16* wih   = (_Float16*)(ws + WIH_OFF);
    _Float16* whh   = (_Float16*)(ws + WHH_OFF);
    _Float16* wlin  = (_Float16*)(ws + WLIN_OFF);
    float*    bias  = (float*)(ws + BIAS_OFF);
    float*    blinp = (float*)(ws + BLIN_OFF);
    float*    ms    = (float*)(ws + MS_OFF);
    float*    rp    = (float*)(ws + RP_OFF);
    unsigned* cnt   = (unsigned*)(ws + CNT_OFF);

    k_prep_x<<<4096, 256, 0, stream>>>(x, x16, B_ * N_ * D_);
    k_prep_w<<<512, 256, 0, stream>>>(Wih, Whh, Wlin, bih, bhh, blin, wih, whh, wlin,
                                      bias, blinp, cnt);
    k_main<<<16 * G4, TPB, 0, stream>>>(x16, wih, whh, wlin, bias, blinp, ms, rp, cnt,
                                        (float*)d_out);
}
